// GATv2Encoder_59768764891991
// MI455X (gfx1250) — matrix operand order repaired and checked by reading, not compiled, not run on hardware
//
#include <hip/hip_runtime.h>
#include <math.h>

typedef __attribute__((ext_vector_type(16))) __bf16 v16bf;
typedef __attribute__((ext_vector_type(8)))  float  v8f;
typedef __attribute__((ext_vector_type(8)))  unsigned int v8u;

#define NNODES 50000
#define NEDGES 400000
#define ETOT   (NEDGES + NNODES)

// ---------- helpers ----------
static __device__ __forceinline__ unsigned short f2bf(float f) {
    unsigned u = __float_as_uint(f);
    unsigned r = u + 0x7FFFu + ((u >> 16) & 1u);   // round-to-nearest-even
    return (unsigned short)(r >> 16);
}

static __device__ __forceinline__ void atomicMaxF(float* addr, float val) {
    if (val >= 0.f) atomicMax((int*)addr, __float_as_int(val));
    else            atomicMin((unsigned int*)addr, __float_as_uint(val));
}

// ---------- pack [Wl;Wr] -> bf16, [bl;br] -> f32 ----------
__global__ void pack_wb_k(const float* __restrict__ Wl, const float* __restrict__ bl,
                          const float* __restrict__ Wr, const float* __restrict__ br,
                          unsigned short* __restrict__ Wc, float* __restrict__ bc,
                          int hc, int Din) {
    int idx = blockIdx.x * blockDim.x + threadIdx.x;
    int total = 2 * hc * Din;
    if (idx < total) {
        int j = idx / Din, k = idx - j * Din;
        float v = (j < hc) ? Wl[j * Din + k] : Wr[(j - hc) * Din + k];
        Wc[idx] = f2bf(v);
    }
    if (idx < 2 * hc) bc[idx] = (idx < hc) ? bl[idx] : br[idx - hc];
}

// ---------- fp32 -> bf16 activation convert ----------
__global__ void cvt_bf16_k(const float* __restrict__ src, unsigned short* __restrict__ dst, int n) {
    int i = blockIdx.x * blockDim.x + threadIdx.x;
    if (i < n) dst[i] = f2bf(src[i]);
}

// ---------- WMMA GEMM: C[M,J] = A[M,K](bf16) * W[J,K](bf16)^T + bias[J] ----------
#define TM 128
#define TN 64
#define TK 32
#define LDA 40
#define LDB 40

__global__ __launch_bounds__(256)
void gemm_bf16_wmma_k(const unsigned short* __restrict__ A,
                      const unsigned short* __restrict__ W,
                      const float* __restrict__ bias,
                      float* __restrict__ Cmat,
                      int M, int K, int J) {
    __shared__ unsigned short As[TM * LDA];
    __shared__ unsigned short Bs[TN * LDB];

    const int tid  = threadIdx.x;
    const int lane = tid & 31;
    const int wave = tid >> 5;
    const int half = lane >> 4;
    const int l16  = lane & 15;
    const int m0   = (wave & 3) * 32;   // wave row offset within block tile
    const int n0   = (wave >> 2) * 32;  // wave col offset within block tile
    const int blockM = blockIdx.x * TM;
    const int blockJ = blockIdx.y * TN;

    v8f acc[2][2] = {};

    for (int k0 = 0; k0 < K; k0 += TK) {
        // stage A tile (128x32 bf16) via 16B chunks
        for (int c = tid; c < (TM * TK / 8); c += 256) {
            int r  = c >> 2;
            int cc = (c & 3) * 8;
            uint4 v = make_uint4(0u, 0u, 0u, 0u);
            int gm = blockM + r;
            if (gm < M) v = *(const uint4*)(A + (size_t)gm * K + k0 + cc);
            *(uint4*)(As + r * LDA + cc) = v;
        }
        // stage B tile (64x32 bf16), natural [j][k] layout (J multiple of 64)
        {
            int c  = tid;                 // exactly 256 chunks
            int r  = c >> 2;
            int cc = (c & 3) * 8;
            uint4 v = *(const uint4*)(W + (size_t)(blockJ + r) * K + k0 + cc);
            *(uint4*)(Bs + r * LDB + cc) = v;
        }
        __syncthreads();

        // A fragments: lane half h holds K = h*8..h*8+7 (v0-3) and 16+h*8.. (v4-7), row M=lane%16
        v16bf afrag[2], bfrag[2];
#pragma unroll
        for (int mt = 0; mt < 2; ++mt) {
            const unsigned short* p = As + (m0 + mt * 16 + l16) * LDA;
            uint4 lo = *(const uint4*)(p + half * 8);
            uint4 hi = *(const uint4*)(p + 16 + half * 8);
            v8u u; u[0] = lo.x; u[1] = lo.y; u[2] = lo.z; u[3] = lo.w;
                   u[4] = hi.x; u[5] = hi.y; u[6] = hi.z; u[7] = hi.w;
            afrag[mt] = __builtin_bit_cast(v16bf, u);
        }
        // B fragments: lane half h holds K = h*8..h*8+7 (v0-3) and 16+h*8.. (v4-7) of column N=lane%16
#pragma unroll
        for (int nt = 0; nt < 2; ++nt) {
            const unsigned short* p = Bs + (n0 + nt * 16 + l16) * LDB;
            uint4 lo = *(const uint4*)(p + half * 8);
            uint4 hi = *(const uint4*)(p + 16 + half * 8);
            v8u u; u[0] = lo.x; u[1] = lo.y; u[2] = lo.z; u[3] = lo.w;
                   u[4] = hi.x; u[5] = hi.y; u[6] = hi.z; u[7] = hi.w;
            bfrag[nt] = __builtin_bit_cast(v16bf, u);
        }
#pragma unroll
        for (int mt = 0; mt < 2; ++mt)
#pragma unroll
            for (int nt = 0; nt < 2; ++nt)
                acc[mt][nt] = __builtin_amdgcn_wmma_f32_16x16x32_bf16(
                    false, afrag[mt], false, bfrag[nt], (short)0, acc[mt][nt], false, false);
        __syncthreads();
    }

    // store D: VGPR v -> row v + half*8, col lane%16
#pragma unroll
    for (int mt = 0; mt < 2; ++mt) {
#pragma unroll
        for (int nt = 0; nt < 2; ++nt) {
            int col = blockJ + n0 + nt * 16 + l16;
            float b = bias[col];
#pragma unroll
            for (int v = 0; v < 8; ++v) {
                int row = blockM + m0 + mt * 16 + v + half * 8;
                if (row < M) Cmat[(size_t)row * J + col] = acc[mt][nt][v] + b;
            }
        }
    }
}

// ---------- softmax state init ----------
__global__ void init_softmax_k(float* __restrict__ m, float* __restrict__ den,
                               float* __restrict__ outb, int nmh, int nout) {
    int i = blockIdx.x * blockDim.x + threadIdx.x;
    if (i < nmh) { m[i] = -INFINITY; den[i] = 0.f; }
    if (i < nout) outb[i] = 0.f;
}

// ---------- pass 1: logits + segment max ----------
__global__ void edge_logits_k(const float* __restrict__ xlr, const float* __restrict__ att,
                              const int* __restrict__ ei, float* __restrict__ logits,
                              float* __restrict__ mbuf,
                              int n, int H, int C, int hc, int J) {
    int idx = blockIdx.x * blockDim.x + threadIdx.x;
    if (idx >= n) return;
    int e = idx / H, h = idx - e * H;
    int s, d;
    if (e < NEDGES) { s = ei[e]; d = ei[NEDGES + e]; } else { s = e - NEDGES; d = s; }
    const float* pl = xlr + (size_t)s * J + h * C;
    const float* pr = xlr + (size_t)d * J + hc + h * C;
    const float* pa = att + h * C;
    float acc = 0.f;
    for (int c = 0; c < C; c += 4) {
        float4 a = *(const float4*)(pl + c);
        float4 b = *(const float4*)(pr + c);
        float4 w = *(const float4*)(pa + c);
        float v;
        v = a.x + b.x; v = v > 0.f ? v : 0.2f * v; acc += w.x * v;
        v = a.y + b.y; v = v > 0.f ? v : 0.2f * v; acc += w.y * v;
        v = a.z + b.z; v = v > 0.f ? v : 0.2f * v; acc += w.z * v;
        v = a.w + b.w; v = v > 0.f ? v : 0.2f * v; acc += w.w * v;
    }
    logits[idx] = acc;
    atomicMaxF(mbuf + (size_t)d * H + h, acc);
}

// ---------- pass 2: exp + segment sum ----------
__global__ void edge_exp_k(float* __restrict__ logits, const float* __restrict__ mbuf,
                           float* __restrict__ den, const int* __restrict__ ei, int n, int H) {
    int idx = blockIdx.x * blockDim.x + threadIdx.x;
    if (idx >= n) return;
    int e = idx / H, h = idx - e * H;
    int d = (e < NEDGES) ? ei[NEDGES + e] : (e - NEDGES);
    float p = __expf(logits[idx] - mbuf[(size_t)d * H + h]);
    logits[idx] = p;
    atomicAdd(den + (size_t)d * H + h, p);
}

// ---------- pass 3: alpha-weighted scatter of xl[src] ----------
__global__ void edge_scatter_k(const float* __restrict__ logits, const float* __restrict__ den,
                               const float* __restrict__ xlr, float* __restrict__ outb,
                               const int* __restrict__ ei, int n, int H, int C, int hc, int J) {
    int idx = blockIdx.x * blockDim.x + threadIdx.x;
    if (idx >= n) return;
    int e = idx / H, h = idx - e * H;
    int s, d;
    if (e < NEDGES) { s = ei[e]; d = ei[NEDGES + e]; } else { s = e - NEDGES; d = s; }
    float alpha = logits[idx] / den[(size_t)d * H + h];
    const float* pl = xlr + (size_t)s * J + h * C;
    float* po = outb + (size_t)d * hc + h * C;
    for (int c = 0; c < C; ++c) atomicAdd(po + c, alpha * pl[c]);
}

// ---------- finalize: + bias, (elu), (residual) ----------
__global__ void finalize_k(const float* __restrict__ outb, const float* __restrict__ bias,
                           const float* __restrict__ hprev, float* __restrict__ dst,
                           int n, int hc, int do_elu, int do_res) {
    int idx = blockIdx.x * blockDim.x + threadIdx.x;
    if (idx >= n) return;
    int j = idx % hc;
    float v = outb[idx] + bias[j];
    if (do_elu) v = v > 0.f ? v : (__expf(v) - 1.f);
    if (do_res) v += hprev[idx];
    dst[idx] = v;
}

extern "C" void kernel_launch(void* const* d_in, const int* in_sizes, int n_in,
                              void* d_out, int out_size, void* d_ws, size_t ws_size,
                              hipStream_t stream) {
    const float* x  = (const float*)d_in[0];
    const int*   ei = (const int*)d_in[1];

    char* ws = (char*)d_ws;
    size_t off = 0;
    auto take = [&](size_t bytes) -> char* {
        char* p = ws + off;
        off += (bytes + 255) & ~(size_t)255;
        return p;
    };
    float*          hbuf   = (float*)take((size_t)NNODES * 256 * 4);
    unsigned short* hbf    = (unsigned short*)take((size_t)NNODES * 256 * 2);
    float*          xlr    = (float*)take((size_t)NNODES * 512 * 4);
    float*          outb   = (float*)take((size_t)NNODES * 256 * 4);
    float*          logits = (float*)take((size_t)ETOT * 8 * 4);
    float*          mbuf   = (float*)take((size_t)NNODES * 8 * 4);
    float*          den    = (float*)take((size_t)NNODES * 8 * 4);
    unsigned short* wcbf   = (unsigned short*)take((size_t)512 * 256 * 2);
    float*          bc     = (float*)take((size_t)512 * 4);
    (void)ws_size; (void)n_in; (void)in_sizes; (void)out_size;

    const int Din_[4] = {128, 256, 256, 256};
    const int H_[4]   = {8, 8, 8, 1};
    const int C_[4]   = {32, 32, 32, 128};

    for (int L = 0; L < 4; ++L) {
        const int Din = Din_[L], H = H_[L], C = C_[L];
        const int hc = H * C, J = 2 * hc;
        const float* Wl   = (const float*)d_in[2 + L * 6 + 0];
        const float* bl   = (const float*)d_in[2 + L * 6 + 1];
        const float* Wr   = (const float*)d_in[2 + L * 6 + 2];
        const float* br   = (const float*)d_in[2 + L * 6 + 3];
        const float* att  = (const float*)d_in[2 + L * 6 + 4];
        const float* bfin = (const float*)d_in[2 + L * 6 + 5];

        int tot = J * Din;
        pack_wb_k<<<(tot + 255) / 256, 256, 0, stream>>>(Wl, bl, Wr, br, wcbf, bc, hc, Din);

        const float* src = (L == 0) ? x : hbuf;
        int nel = NNODES * Din;
        cvt_bf16_k<<<(nel + 255) / 256, 256, 0, stream>>>(src, hbf, nel);

        dim3 grid((NNODES + TM - 1) / TM, J / TN);
        gemm_bf16_wmma_k<<<grid, 256, 0, stream>>>(hbf, wcbf, bc, xlr, NNODES, Din, J);

        int ninit = NNODES * hc;
        init_softmax_k<<<(ninit + 255) / 256, 256, 0, stream>>>(mbuf, den, outb, NNODES * H, ninit);

        int ne = ETOT * H;
        edge_logits_k<<<(ne + 255) / 256, 256, 0, stream>>>(xlr, att, ei, logits, mbuf, ne, H, C, hc, J);
        edge_exp_k<<<(ne + 255) / 256, 256, 0, stream>>>(logits, mbuf, den, ei, ne, H);
        edge_scatter_k<<<(ne + 255) / 256, 256, 0, stream>>>(logits, den, xlr, outb, ei, ne, H, C, hc, J);

        float* dst = (L == 3) ? (float*)d_out : hbuf;
        finalize_k<<<(ninit + 255) / 256, 256, 0, stream>>>(outb, bfin, hbuf, dst, ninit, hc,
                                                            (L < 3) ? 1 : 0,
                                                            (L == 1 || L == 2) ? 1 : 0);
    }
}
